// SimplifiedMambaSSM_21964462752066
// MI455X (gfx1250) — matrix operand order repaired and checked by reading, not compiled, not run on hardware
//
#include <hip/hip_runtime.h>
#include <hip/hip_bf16.h>
#include <math.h>

#define D_MODELX 1024
#define D_STATEX 16
#define BATCHX   8
#define SEQX     4096
#define NROWSX   (BATCHX * SEQX)   // 32768
#define LN_EPSX  1e-5f

typedef __attribute__((ext_vector_type(16))) __bf16 v16bf;
typedef __attribute__((ext_vector_type(8)))  float  v8f;

union Frag32B { uint4 u[2]; v16bf v; };

// ---------------------------------------------------------------------------
// Kernel 1: pack W_gate (1024x1024) and [W_delta;W_B;W_C] (48x1024) to bf16.
// ---------------------------------------------------------------------------
__global__ void __launch_bounds__(256)
pack_weights_kernel(const float* __restrict__ Wg,
                    const float* __restrict__ Wd,
                    const float* __restrict__ Wb,
                    const float* __restrict__ Wc,
                    __bf16* __restrict__ wgb,
                    __bf16* __restrict__ Pb) {
  const int idx = blockIdx.x * 256 + threadIdx.x;
  const int NG = D_MODELX * D_MODELX;
  if (idx < NG) {
    wgb[idx] = (__bf16)Wg[idx];
  } else {
    const int j = idx - NG;
    if (j < 48 * D_MODELX) {
      const int r = j >> 10, c = j & (D_MODELX - 1);
      float v = (r < 16) ? Wd[r * D_MODELX + c]
              : (r < 32) ? Wb[(r - 16) * D_MODELX + c]
                         : Wc[(r - 32) * D_MODELX + c];
      Pb[j] = (__bf16)v;
    }
  }
}

// ---------------------------------------------------------------------------
// WMMA fragment loaders (CDNA5 16x16x32 bf16 layouts, ISA 7.12.2).
// A tile (16 rows x 32 K), row-major source, stride ld (bf16 elems):
//   lane L: row = L&15, hi = L>>4; VGPR0-3 <- K[8*hi .. 8*hi+7],
//   VGPR4-7 <- K[16+8*hi .. 16+8*hi+7]  ==> two b128 loads.
// B tile from [N,K] row-major weight (i.e. column n of B is weight row n):
//   lane L: n = L&15, khi = L>>4; VGPR0-3 <- K[8*khi .. 8*khi+7],
//   VGPR4-7 <- K[16+8*khi .. 16+8*khi+7]  ==> two b128 loads.
// ---------------------------------------------------------------------------
__device__ __forceinline__ v16bf load_frag_a(const __bf16* base, int ld,
                                             int lane, int k0) {
  const int r = lane & 15, hi = lane >> 4;
  const __bf16* p = base + (size_t)r * ld + k0 + hi * 8;
  Frag32B f;
  f.u[0] = *(const uint4*)(p);
  f.u[1] = *(const uint4*)(p + 16);
  return f.v;
}

__device__ __forceinline__ v16bf load_frag_b(const __bf16* base, int ld,
                                             int lane, int k0) {
  const int n = lane & 15, khi = lane >> 4;
  const __bf16* p = base + (size_t)n * ld + k0 + khi * 8;
  Frag32B f;
  f.u[0] = *(const uint4*)(p);
  f.u[1] = *(const uint4*)(p + 16);
  return f.v;
}

// ---------------------------------------------------------------------------
// Kernel 2: per 32-row tile:
//   G = x @ Wg^T (bf16 WMMA, f32 acc)  -> LDS
//   gate = sigmoid(LN(G + b_gate))     -> xg = gate * x (overwrites A tile)
//   S = xg @ [Wd;Wb;Wc]^T              -> A_bar / Bt / Ct to global ws
// ---------------------------------------------------------------------------
#define MTX  32
#define ALDX 1032   // bf16 row stride (16B aligned: 1032*2 = 2064)
#define GLDX 1028   // f32 row stride

__global__ void __launch_bounds__(256)
gate_proj_kernel(const float* __restrict__ x,
                 const float* __restrict__ b_gate,
                 const float* __restrict__ ln_w,
                 const float* __restrict__ ln_b,
                 const float* __restrict__ A_diag,
                 const __bf16* __restrict__ wgb,
                 const __bf16* __restrict__ Pb,
                 float* __restrict__ abar,
                 float* __restrict__ btv,
                 float* __restrict__ ctv) {
  extern __shared__ char smem[];
  __bf16* As = (__bf16*)smem;                                   // [MTX][ALDX]
  float*  Gs = (float*)(smem + MTX * ALDX * sizeof(__bf16));    // [MTX][GLDX]

  const int tid  = threadIdx.x;
  const int lane = tid & 31;
  const int wave = tid >> 5;
  const int hi   = lane >> 4;
  const size_t row0 = (size_t)blockIdx.x * MTX;

  // Stage x tile (f32 -> bf16) into LDS.
  for (int i = tid; i < MTX * D_MODELX; i += 256) {
    const int r = i >> 10, c = i & (D_MODELX - 1);
    As[r * ALDX + c] = (__bf16)x[(row0 + r) * D_MODELX + c];
  }
  __syncthreads();

  // ---- GEMM: 32x1024 = (32x1024) x (1024x1024), wave grid 2(M) x 4(N) ----
  const int wm = wave >> 2;   // 0..1 : 16-row slab
  const int wn = wave & 3;    // 0..3 : 32-col slab within a 128-col chunk
  const __bf16* Abase = As + (size_t)(16 * wm) * ALDX;

  for (int nc = 0; nc < 8; ++nc) {
    const int c0 = nc * 128 + 32 * wn;
    v8f acc0 = {0.f, 0.f, 0.f, 0.f, 0.f, 0.f, 0.f, 0.f};
    v8f acc1 = {0.f, 0.f, 0.f, 0.f, 0.f, 0.f, 0.f, 0.f};
    const __bf16* B0 = wgb + (size_t)c0 * D_MODELX;
    const __bf16* B1 = wgb + (size_t)(c0 + 16) * D_MODELX;
#pragma unroll 4
    for (int k0 = 0; k0 < D_MODELX; k0 += 32) {
      v16bf a  = load_frag_a(Abase, ALDX, lane, k0);
      v16bf b0 = load_frag_b(B0, D_MODELX, lane, k0);
      v16bf b1 = load_frag_b(B1, D_MODELX, lane, k0);
      acc0 = __builtin_amdgcn_wmma_f32_16x16x32_bf16(false, a, false, b0,
                                                     (short)0, acc0, false, false);
      acc1 = __builtin_amdgcn_wmma_f32_16x16x32_bf16(false, a, false, b1,
                                                     (short)0, acc1, false, false);
    }
    const int n = lane & 15;
#pragma unroll
    for (int v = 0; v < 8; ++v) {
      const int m = 16 * wm + v + 8 * hi;     // C/D layout: vgpr v, lane half
      Gs[m * GLDX + c0 + n]      = acc0[v];
      Gs[m * GLDX + c0 + 16 + n] = acc1[v];
    }
  }
  __syncthreads();

  // ---- LayerNorm(G + b_gate) -> sigmoid -> xg = gate * x (4 rows/wave) ----
  for (int i = 0; i < 4; ++i) {
    const int r = wave * 4 + i;
    float sum = 0.f, ss = 0.f;
    for (int c = lane; c < D_MODELX; c += 32) {
      const float v = Gs[r * GLDX + c] + b_gate[c];
      sum += v; ss += v * v;
    }
#pragma unroll
    for (int off = 16; off > 0; off >>= 1) {
      sum += __shfl_xor(sum, off, 32);
      ss  += __shfl_xor(ss,  off, 32);
    }
    const float mu   = sum * (1.f / D_MODELX);
    const float var  = ss * (1.f / D_MODELX) - mu * mu;
    const float rstd = rsqrtf(var + LN_EPSX);
    for (int c = lane; c < D_MODELX; c += 32) {
      const float v    = Gs[r * GLDX + c] + b_gate[c];
      const float ln   = (v - mu) * rstd * ln_w[c] + ln_b[c];
      const float gate = 1.f / (1.f + __expf(-ln));
      const float xv   = (float)As[r * ALDX + c];
      As[r * ALDX + c] = (__bf16)(gate * xv);
    }
  }
  __syncthreads();

  // ---- Projections: S = xg @ P^T  (32 x 48), 6 tiles on waves 0..5 ----
  if (wave < 6) {
    const int mi = wave & 1;      // row slab
    const int pj = wave >> 1;     // 0: delta, 1: B, 2: C
    const __bf16* Ab = As + (size_t)(16 * mi) * ALDX;
    const __bf16* Bb = Pb + (size_t)(16 * pj) * D_MODELX;
    v8f acc = {0.f, 0.f, 0.f, 0.f, 0.f, 0.f, 0.f, 0.f};
#pragma unroll 4
    for (int k0 = 0; k0 < D_MODELX; k0 += 32) {
      v16bf a = load_frag_a(Ab, ALDX, lane, k0);
      v16bf b = load_frag_b(Bb, D_MODELX, lane, k0);
      acc = __builtin_amdgcn_wmma_f32_16x16x32_bf16(false, a, false, b,
                                                    (short)0, acc, false, false);
    }
    const int s = lane & 15;
    const float Ad = A_diag[s];
#pragma unroll
    for (int v = 0; v < 8; ++v) {
      const size_t row = row0 + 16 * mi + v + 8 * hi;
      const float val = acc[v];
      if (pj == 0) {            // A_bar = exp(softplus(val) * A_diag[s])
        const float d = (val > 20.f) ? val : log1pf(__expf(val));
        abar[row * D_STATEX + s] = __expf(d * Ad);
      } else if (pj == 1) {
        btv[row * D_STATEX + s] = val;
      } else {
        ctv[row * D_STATEX + s] = val;
      }
    }
  }
}

// ---------------------------------------------------------------------------
// Kernel 3: sequential scan, one block per sequence.
//   wave0 lanes 0-15: recurrence h = tanh(a*h+b), masked; emit sv = Ct*h.
//   waves 1-7: double-buffered prefetch of next 128-step chunk into LDS.
// ---------------------------------------------------------------------------
#define SCHX 128

__global__ void __launch_bounds__(256)
scan_kernel(const float* __restrict__ abar,
            const float* __restrict__ btv,
            const float* __restrict__ ctv,
            const float* __restrict__ mask,
            const float* __restrict__ h0,
            float* __restrict__ sv,
            float* __restrict__ hfin) {
  __shared__ float aS[2][SCHX * D_STATEX];
  __shared__ float bS[2][SCHX * D_STATEX];
  __shared__ float cS[2][SCHX * D_STATEX];
  __shared__ float mS[2][SCHX];

  const int tid = threadIdx.x;
  const int b = blockIdx.x;
  const size_t seqbase = (size_t)b * SEQX;

  // Preload chunk 0 (all threads).
  {
    const size_t g0 = seqbase * D_STATEX;
    for (int i = tid; i < SCHX * D_STATEX; i += 256) {
      aS[0][i] = abar[g0 + i]; bS[0][i] = btv[g0 + i]; cS[0][i] = ctv[g0 + i];
    }
    for (int i = tid; i < SCHX; i += 256) mS[0][i] = mask[seqbase + i];
  }
  __syncthreads();

  float h = (tid < D_STATEX) ? h0[b * D_STATEX + tid] : 0.f;

  const int NCHUNK = SEQX / SCHX;   // 32
  for (int c = 0; c < NCHUNK; ++c) {
    const int buf = c & 1;
    if (tid < 16) {
      // scan this chunk (lanes 16-31 of wave 0 idle -> no serialization
      // with loader waves)
      const int s = tid;
      size_t out = (seqbase + (size_t)c * SCHX) * D_STATEX + s;
#pragma unroll 4
      for (int j = 0; j < SCHX; ++j) {
        const int idx = j * D_STATEX + s;
        const float hc = tanhf(fmaf(aS[buf][idx], h, bS[buf][idx]));
        const float m  = mS[buf][j];
        h = fmaf(m, hc - h, h);            // hc*m + h*(1-m)
        sv[out] = cS[buf][idx] * h;
        out += D_STATEX;
      }
    } else if (tid >= 32 && (c + 1) < NCHUNK) {
      // waves 1-7 prefetch chunk c+1
      const int nb = buf ^ 1;
      const size_t gb = (seqbase + (size_t)(c + 1) * SCHX) * D_STATEX;
      for (int i = tid - 32; i < SCHX * D_STATEX; i += 224) {
        aS[nb][i] = abar[gb + i]; bS[nb][i] = btv[gb + i]; cS[nb][i] = ctv[gb + i];
      }
      for (int i = tid - 32; i < SCHX; i += 224)
        mS[nb][i] = mask[seqbase + (size_t)(c + 1) * SCHX + i];
    }
    __syncthreads();
  }
  if (tid < D_STATEX) hfin[b * D_STATEX + tid] = h;
}

// ---------------------------------------------------------------------------
// Kernel 4: y = sv @ W_out^T + D * x   (memory-bound epilogue, K=16)
// ---------------------------------------------------------------------------
__global__ void __launch_bounds__(256)
out_kernel(const float* __restrict__ sv,
           const float* __restrict__ x,
           const float* __restrict__ Wout,
           const float* __restrict__ Dv,
           float* __restrict__ y) {
  __shared__ float svs[D_STATEX];
  const size_t row = blockIdx.x;
  const int tid = threadIdx.x;
  if (tid < D_STATEX) svs[tid] = sv[row * D_STATEX + tid];
  __syncthreads();
#pragma unroll
  for (int j = 0; j < 4; ++j) {
    const int d = tid + j * 256;
    const float* w = Wout + (size_t)d * D_STATEX;
    float acc = 0.f;
#pragma unroll
    for (int s = 0; s < D_STATEX; ++s) acc = fmaf(svs[s], w[s], acc);
    y[row * D_MODELX + d] = acc + Dv[d] * x[row * D_MODELX + d];
  }
}

// ---------------------------------------------------------------------------
extern "C" void kernel_launch(void* const* d_in, const int* in_sizes, int n_in,
                              void* d_out, int out_size, void* d_ws, size_t ws_size,
                              hipStream_t stream) {
  (void)in_sizes; (void)n_in; (void)out_size; (void)ws_size;

  const float* x       = (const float*)d_in[0];
  const float* h0      = (const float*)d_in[1];
  const float* mask    = (const float*)d_in[2];
  const float* A_diag  = (const float*)d_in[3];
  const float* W_delta = (const float*)d_in[4];
  const float* W_B     = (const float*)d_in[5];
  const float* W_C     = (const float*)d_in[6];
  const float* W_out   = (const float*)d_in[7];
  const float* Dv      = (const float*)d_in[8];
  const float* W_gate  = (const float*)d_in[9];
  const float* b_gate  = (const float*)d_in[10];
  const float* ln_w    = (const float*)d_in[11];
  const float* ln_b    = (const float*)d_in[12];

  float* y    = (float*)d_out;
  float* hfin = y + (size_t)NROWSX * D_MODELX;

  // workspace layout (~10.6 MB)
  __bf16* wgb = (__bf16*)d_ws;                              // 1024*1024 bf16
  __bf16* Pb  = wgb + (size_t)D_MODELX * D_MODELX;          // 48*1024 bf16
  float* abar = (float*)(Pb + 48 * D_MODELX);               // 32768*16 f32
  float* btv  = abar + (size_t)NROWSX * D_STATEX;
  float* ctv  = btv  + (size_t)NROWSX * D_STATEX;
  float* sv   = ctv  + (size_t)NROWSX * D_STATEX;

  {
    const int total = D_MODELX * D_MODELX + 48 * D_MODELX;
    pack_weights_kernel<<<(total + 255) / 256, 256, 0, stream>>>(
        W_gate, W_delta, W_B, W_C, wgb, Pb);
  }
  {
    const size_t smem = (size_t)MTX * ALDX * sizeof(__bf16) +
                        (size_t)MTX * GLDX * sizeof(float);   // ~197.6 KB < 320 KB
    gate_proj_kernel<<<NROWSX / MTX, 256, smem, stream>>>(
        x, b_gate, ln_w, ln_b, A_diag, wgb, Pb, abar, btv, ctv);
  }
  scan_kernel<<<BATCHX, 256, 0, stream>>>(abar, btv, ctv, mask, h0, sv, hfin);
  out_kernel<<<NROWSX, 256, 0, stream>>>(sv, x, W_out, Dv, y);
}
